// DiscreteCommPolicy_60352880443771
// MI455X (gfx1250) — matrix operand order repaired and checked by reading, not compiled, not run on hardware
//
#include <hip/hip_runtime.h>
#include <hip/hip_bf16.h>

#define BB 65536
#define NA 10
#define OBSD 64
#define HH 50
#define SS 10
#define AA 2
#define KK 2
#define INF_ 110
#define NEGV (-1e9f)

typedef __attribute__((ext_vector_type(16))) _Float16 v16h;
typedef __attribute__((ext_vector_type(8)))  _Float16 v8h;
typedef __attribute__((ext_vector_type(8)))  float    v8f;

#define WENC_KST 72    // halves per column (K-stride) for encoder weights
#define WF_KST   136
#define WCOM_KST 72
#define WHD_KST  72
#define INP_ST   136   // halves per row of the 16x128 input tile (272B -> conflict-free)
#define DUMP_ST  17

// Branch-free tanh using the HW transcendental unit: v_exp_f32 + v_rcp_f32.
// Saturates correctly: x>>0 -> exp=inf -> 1-0 = 1 ; x<<0 -> exp=0 -> 1-2 = -1.
__device__ __forceinline__ float tanh_fast(float x) {
  float e = __expf(2.f * x);
  return 1.f - 2.f * __builtin_amdgcn_rcpf(e + 1.f);
}

__device__ __forceinline__ v16h make16(v8h lo, v8h hi) {
  return __builtin_shufflevector(lo, hi, 0,1,2,3,4,5,6,7,8,9,10,11,12,13,14,15);
}

// A-fragment (16x32 f16) from an LDS tile laid out [16 rows][INP_ST halves].
// ISA layout: lane m=L%16 is row M; lane group g=L/16 selects K-half:
//   a[i], i<8  -> K = kbase + g*8 + i
//   a[i], i>=8 -> K = kbase + 16 + g*8 + (i-8)
__device__ __forceinline__ v16h load_a_lds(const _Float16* tile, int lane, int kbase) {
  const int m = lane & 15, g = lane >> 4;
  const _Float16* p = tile + m * INP_ST + kbase + g * 8;
  v8h lo = *(const v8h*)p;
  v8h hi = *(const v8h*)(p + 16);
  return make16(lo, hi);
}

// B-fragment (32x16 f16) from transposed LDS weights WT[col][k] with K-stride kst.
// ISA layout: lane n=L%16 is column N; b[i], i<8 -> K = kbase + g*8 + i; i>=8 -> K = kbase + 16 + g*8 + (i-8)
__device__ __forceinline__ v16h load_b_lds(const _Float16* WT, int kst, int lane, int kbase) {
  const int n = lane & 15, g = lane >> 4;
  const _Float16* p = WT + n * kst + kbase + g * 8;
  v8h lo = *(const v8h*)p;
  v8h hi = *(const v8h*)(p + 16);
  return make16(lo, hi);
}

// A-fragment straight from global fp32 obs rows (row stride OBSD floats).
__device__ __forceinline__ v16h load_a_obs(const float* base, int lane, int kbase) {
  const int m = lane & 15, g = lane >> 4;
  const float* p = base + (size_t)m * OBSD + kbase + g * 8;
  float4 x0 = *(const float4*)(p);
  float4 x1 = *(const float4*)(p + 4);
  float4 y0 = *(const float4*)(p + 16);
  float4 y1 = *(const float4*)(p + 20);
  v16h a;
  a[0]=(_Float16)x0.x; a[1]=(_Float16)x0.y; a[2]=(_Float16)x0.z; a[3]=(_Float16)x0.w;
  a[4]=(_Float16)x1.x; a[5]=(_Float16)x1.y; a[6]=(_Float16)x1.z; a[7]=(_Float16)x1.w;
  a[8]=(_Float16)y0.x; a[9]=(_Float16)y0.y; a[10]=(_Float16)y0.z; a[11]=(_Float16)y0.w;
  a[12]=(_Float16)y1.x; a[13]=(_Float16)y1.y; a[14]=(_Float16)y1.z; a[15]=(_Float16)y1.w;
  return a;
}

#define WMMA(A,B,C) __builtin_amdgcn_wmma_f32_16x16x32_f16(false, (A), false, (B), (short)0, (C), false, false)

__global__ __launch_bounds__(320) void DiscreteCommPolicy_kernel(
    const float* __restrict__ obs,   const float* __restrict__ mask,
    const float* __restrict__ W_enc, const float* __restrict__ b_enc,
    const float* __restrict__ W_f,   const float* __restrict__ b_f,
    const float* __restrict__ W_comm,const float* __restrict__ b_comm,
    const float* __restrict__ W_pi,  const float* __restrict__ b_pi,
    const float* __restrict__ W_v,   const float* __restrict__ b_v,
    float* __restrict__ out)
{
  __shared__ _Float16 sWencT[64 * WENC_KST];     // [n][k], n pad 50->64, K=64
  __shared__ _Float16 sWfT[2 * 64 * WF_KST];     // [step][n][k], K pad 110->128
  __shared__ _Float16 sWcomT[16 * WCOM_KST];     // [n][k], n pad 10->16, K pad 50->64 (zeros)
  __shared__ _Float16 sWhdT[16 * WHD_KST];       // cols 0,1 = W_pi, col 2 = W_v
  __shared__ float sBenc[64], sBf[2 * 64], sBcom[16], sBhd[16];
  __shared__ _Float16 sInp[10 * 16 * INP_ST];    // per-wave 16x128 concat tile [h|c|h0|0]
  __shared__ float sDump[10 * 16 * DUMP_ST];     // per-wave 16x16 comm-logit dump
  __shared__ unsigned sCbits[2 * 16];            // double-buffered per-batch symbol bitmask

  const int tid  = threadIdx.x;
  const int wave = tid >> 5;
  const int lane = tid & 31;
  const int g    = lane >> 4;

  // ---- convert + transpose weights into LDS (zero-padded) ----
  for (int idx = tid; idx < 64 * WENC_KST; idx += 320) {
    int n = idx / WENC_KST, k = idx % WENC_KST;
    sWencT[idx] = (_Float16)((n < HH && k < OBSD) ? W_enc[k * HH + n] : 0.f);
  }
  for (int idx = tid; idx < 2 * 64 * WF_KST; idx += 320) {
    int i = idx / (64 * WF_KST);
    int r = idx % (64 * WF_KST);
    int n = r / WF_KST, k = r % WF_KST;
    sWfT[idx] = (_Float16)((n < HH && k < INF_) ? W_f[((size_t)i * INF_ + k) * HH + n] : 0.f);
  }
  for (int idx = tid; idx < 16 * WCOM_KST; idx += 320) {
    int n = idx / WCOM_KST, k = idx % WCOM_KST;
    sWcomT[idx] = (_Float16)((n < SS && k < HH) ? W_comm[k * SS + n] : 0.f);
  }
  for (int idx = tid; idx < 16 * WHD_KST; idx += 320) {
    int n = idx / WHD_KST, k = idx % WHD_KST;
    float v = 0.f;
    if (k < HH) { if (n < AA) v = W_pi[k * AA + n]; else if (n == AA) v = W_v[k]; }
    sWhdT[idx] = (_Float16)v;
  }
  if (tid < 64)  sBenc[tid] = (tid < HH) ? b_enc[tid] : 0.f;
  if (tid < 128) { int i = tid >> 6, j = tid & 63; sBf[tid] = (j < HH) ? b_f[i * HH + j] : 0.f; }
  if (tid < 16)  sBcom[tid] = (tid < SS) ? b_comm[tid] : 0.f;
  if (tid < 16)  sBhd[tid]  = (tid < AA) ? b_pi[tid] : (tid == AA ? b_v[0] : 0.f);

  _Float16* myInp = sInp + wave * 16 * INP_ST;
  float*    myDump = sDump + wave * 16 * DUMP_ST;

  // zero c region (K 50..59) and tail pad (K 110..127) of this wave's tile
  if (lane < 16) {
    for (int k = HH; k < HH + SS; ++k)  myInp[lane * INP_ST + k] = (_Float16)0.f;
    for (int k = INF_; k < 128; ++k)    myInp[lane * INP_ST + k] = (_Float16)0.f;
  }
  __syncthreads();

  const int tileRow = (blockIdx.x * 10 + wave) * 16;   // base row into flattened [B*N]

  // ---- encoder: h0 = tanh(obs @ W_enc + b) ; h = h0 ----
  {
    v16h aE0 = load_a_obs(obs + (size_t)tileRow * OBSD, lane, 0);
    v16h aE1 = load_a_obs(obs + (size_t)tileRow * OBSD, lane, 32);
    #pragma unroll
    for (int t = 0; t < 4; ++t) {
      v8f acc = {};
      acc = WMMA(aE0, load_b_lds(sWencT + (t * 16) * WENC_KST, WENC_KST, lane, 0),  acc);
      acc = WMMA(aE1, load_b_lds(sWencT + (t * 16) * WENC_KST, WENC_KST, lane, 32), acc);
      const int col = t * 16 + (lane & 15);
      if (col < HH) {
        const float bias = sBenc[col];
        #pragma unroll
        for (int v = 0; v < 8; ++v) {
          _Float16 hv = (_Float16)tanh_fast(acc[v] + bias);
          const int r = v + g * 8;
          myInp[r * INP_ST + col] = hv;        // h region  (K 0..49)
          myInp[r * INP_ST + 60 + col] = hv;   // h0 region (K 60..109)
        }
      }
    }
  }

  float comm_lp = 0.f;
  const float mk = (lane < 16) ? mask[tileRow + lane] : 0.f;

  // ---- K comm steps ----
  for (int step = 0; step < KK; ++step) {
    // comm logits = h @ W_comm (rows 50..63 of W_comm are zero, so reading K 0..63 is safe)
    {
      v16h a0 = load_a_lds(myInp, lane, 0);
      v16h a1 = load_a_lds(myInp, lane, 32);
      v8f acc = {};
      acc = WMMA(a0, load_b_lds(sWcomT, WCOM_KST, lane, 0),  acc);
      acc = WMMA(a1, load_b_lds(sWcomT, WCOM_KST, lane, 32), acc);
      const int n = lane & 15;
      #pragma unroll
      for (int v = 0; v < 8; ++v)
        myDump[(v + g * 8) * DUMP_ST + n] = acc[v];
    }
    unsigned* cb = sCbits + (step & 1) * 16;
    if (tid < 16) cb[tid] = 0u;
    __syncthreads();

    // per-row: masked log-softmax, greedy argmax, bag-of-words OR across agents
    if (lane < 16) {
      float x[SS];
      float mx = -3.4e38f; int w = 0;
      #pragma unroll
      for (int s = 0; s < SS; ++s) {
        float v = myDump[lane * DUMP_ST + s] + sBcom[s];
        if (mk == 0.f) v = NEGV;
        x[s] = v;
        if (v > mx) { mx = v; w = s; }   // first max wins (strict >)
      }
      float sum = 0.f;
      #pragma unroll
      for (int s = 0; s < SS; ++s) sum += __expf(x[s] - mx);
      comm_lp += (-__logf(sum)) * mk;    // logp at argmax = -log(sum)
      const int lb = (wave * 16 + lane) / NA;
      atomicOr(&cb[lb], 1u << w);
    }
    __syncthreads();

    // broadcast c into this tile's K 50..59
    if (lane < 16) {
      const unsigned bits = cb[(wave * 16 + lane) / NA];
      #pragma unroll
      for (int s = 0; s < SS; ++s)
        myInp[lane * INP_ST + HH + s] = (_Float16)(((bits >> s) & 1u) ? 1.f : 0.f);
    }

    // h = tanh([h|c|h0] @ W_f[step] + b_f[step])  (K=110 padded to 128 with zero rows)
    {
      const _Float16* Wf = sWfT + step * 64 * WF_KST;
      const float*    Bf = sBf + step * 64;
      v16h f0 = load_a_lds(myInp, lane, 0);
      v16h f1 = load_a_lds(myInp, lane, 32);
      v16h f2 = load_a_lds(myInp, lane, 64);
      v16h f3 = load_a_lds(myInp, lane, 96);
      #pragma unroll
      for (int t = 0; t < 4; ++t) {
        const _Float16* Wt = Wf + (t * 16) * WF_KST;
        v8f acc = {};
        acc = WMMA(f0, load_b_lds(Wt, WF_KST, lane, 0),  acc);
        acc = WMMA(f1, load_b_lds(Wt, WF_KST, lane, 32), acc);
        acc = WMMA(f2, load_b_lds(Wt, WF_KST, lane, 64), acc);
        acc = WMMA(f3, load_b_lds(Wt, WF_KST, lane, 96), acc);
        const int col = t * 16 + (lane & 15);
        if (col < HH) {
          const float bias = Bf[col];
          #pragma unroll
          for (int v = 0; v < 8; ++v)
            myInp[(v + g * 8) * INP_ST + col] = (_Float16)tanh_fast(acc[v] + bias);
        }
      }
    }
  }

  // ---- heads: [logits | baseline] = h @ [W_pi | W_v] ----
  {
    v16h a0 = load_a_lds(myInp, lane, 0);
    v16h a1 = load_a_lds(myInp, lane, 32);
    v8f acc = {};
    acc = WMMA(a0, load_b_lds(sWhdT, WHD_KST, lane, 0),  acc);
    acc = WMMA(a1, load_b_lds(sWhdT, WHD_KST, lane, 32), acc);

    const size_t offBase = (size_t)BB * NA * AA;       // baseline offset
    const size_t offLogp = offBase + (size_t)BB * NA;  // comm_logp offset
    const size_t offC    = offLogp + (size_t)BB * NA;  // c offset

    const int n = lane & 15;
    if (n <= AA) {
      const float bias = sBhd[n];
      #pragma unroll
      for (int v = 0; v < 8; ++v) {
        const int r = tileRow + v + g * 8;
        const float mrow = mask[r];
        const float val = acc[v] + bias;
        if (n < AA) out[(size_t)r * AA + n] = (mrow == 0.f) ? NEGV : val;
        else        out[offBase + r] = val * mrow;
      }
    }
    if (lane < 16) {
      const int r = tileRow + lane;
      out[offLogp + r] = comm_lp;
      const unsigned bits = sCbits[((KK - 1) & 1) * 16 + (wave * 16 + lane) / NA];
      #pragma unroll
      for (int s = 0; s < SS; ++s)
        out[offC + (size_t)r * SS + s] = ((bits >> s) & 1u) ? 1.f : 0.f;
    }
  }
}

extern "C" void kernel_launch(void* const* d_in, const int* in_sizes, int n_in,
                              void* d_out, int out_size, void* d_ws, size_t ws_size,
                              hipStream_t stream) {
  const float* obs    = (const float*)d_in[0];
  const float* mask   = (const float*)d_in[1];
  const float* W_enc  = (const float*)d_in[2];
  const float* b_enc  = (const float*)d_in[3];
  const float* W_f    = (const float*)d_in[4];
  const float* b_f    = (const float*)d_in[5];
  const float* W_comm = (const float*)d_in[6];
  const float* b_comm = (const float*)d_in[7];
  const float* W_pi   = (const float*)d_in[8];
  const float* b_pi   = (const float*)d_in[9];
  const float* W_v    = (const float*)d_in[10];
  const float* b_v    = (const float*)d_in[11];

  dim3 grid((BB * NA) / 160);   // 4096 workgroups x 16 batches each
  dim3 block(320);              // 10 wave32s, one 16-row WMMA tile each
  DiscreteCommPolicy_kernel<<<grid, block, 0, stream>>>(
      obs, mask, W_enc, b_enc, W_f, b_f, W_comm, b_comm,
      W_pi, b_pi, W_v, b_v, (float*)d_out);
}
